// CrossAttention_85160611545787
// MI455X (gfx1250) — matrix operand order repaired and checked by reading, not compiled, not run on hardware
//
#include <hip/hip_runtime.h>
#include <hip/hip_bf16.h>

// ---------------------------------------------------------------------------
// Problem dimensions (from reference): B=64, T=512, N=8, D=128
// MODEL = 1024, HID = 4096, ROWS = B*T = 32768
// ---------------------------------------------------------------------------
#define BB    64
#define TT    512
#define NH    8
#define DD    128
#define MODEL 1024
#define HID   4096
#define ROWS  (BB * TT)

typedef __bf16 bf16_t;
typedef __bf16 v16bf  __attribute__((ext_vector_type(16)));
typedef __bf16 bf16x8 __attribute__((ext_vector_type(8)));
typedef float  v8f    __attribute__((ext_vector_type(8)));
typedef unsigned int u32x4 __attribute__((ext_vector_type(4)));
typedef int   i32x8 __attribute__((ext_vector_type(8)));
typedef int   i32x4 __attribute__((ext_vector_type(4)));

union AFrag { v16bf v; bf16x8 h[2]; };
union CFrag { v8f v; float f[8]; };

// A fragment (16x32 bf16, row-major source, ISA 7.12.2 layout):
// lane<16 : row = lane,    elems 0..7 = K 0..7,  elems 8..15 = K 16..23
// lane>=16: row = lane-16, elems 0..7 = K 8..15, elems 8..15 = K 24..31
__device__ __forceinline__ v16bf load_a_frag(const bf16_t* __restrict__ base,
                                             int lda, int lane) {
    const int m    = lane & 15;
    const int half = lane >> 4;
    const bf16_t* p = base + (size_t)m * lda + half * 8;
    AFrag u;
    u.h[0] = *(const bf16x8*)(p);
    u.h[1] = *(const bf16x8*)(p + 16);
    return u.v;
}

// B fragment (32x16 bf16): lane holds column lane&15, elems 0..7 = K 8h..8h+7, elems 8..15 = K 16+8h..23+8h.
__device__ __forceinline__ v16bf load_b_frag(const bf16_t* base, int ldb,
                                             int lane) {
    AFrag u;
    const int n = lane & 15, half = lane >> 4;
    const bf16_t* p = base + n;
#pragma unroll
    for (int s = 0; s < 8; ++s) {
        u.v[s] = p[(size_t)(8 * half + s) * ldb];
        u.v[8 + s] = p[(size_t)(16 + 8 * half + s) * ldb];
    }
    return u.v;
}

// ---------------------------------------------------------------------------
// TDM: stage a rows x cols bf16 tile (row-major, stride ld elements) from
// global into LDS. 2D descriptor per CDNA5 ISA ch.8: group0 holds
// count/lds_addr/global_addr/type, group1 holds data_size, tensor/tile dims,
// dim0 stride. Issued by wave 0 only (uniform branch), completion via
// TENSORcnt, published to the workgroup by the barrier at the call site.
// This toolchain exposes the 6-argument builtin form:
//   (u32x4 grp0, i32x8 grp1, i32x4 grp2, i32x4 grp3, i32x8 pad, i32 cpol)
// ---------------------------------------------------------------------------
#if __has_builtin(__builtin_amdgcn_tensor_load_to_lds)
#define USE_TDM 1
#else
#define USE_TDM 0
#endif

#if USE_TDM
__device__ __forceinline__ void tdm_load_2d_bf16(unsigned int lds_addr,
                                                 const void* gsrc,
                                                 unsigned int rows,
                                                 unsigned int cols,
                                                 unsigned int ld) {
    const unsigned long long ga = (unsigned long long)(uintptr_t)gsrc;
    u32x4 g0;
    g0.x = 1u;                                  // count=1, user-mode, no gather
    g0.y = lds_addr;                            // LDS byte address of tile
    g0.z = (unsigned int)(ga & 0xFFFFFFFFu);    // global_addr[31:0]
    g0.w = (unsigned int)((ga >> 32) & 0x01FFFFFFu) | (2u << 30); // addr|type=2
    i32x8 g1;
    g1[0] = (int)(1u << 16);                    // wg_mask=0, data_size=1 (2B)
    g1[1] = (int)((cols & 0xFFFFu) << 16);      // tensor_dim0 lo -> bits 63:48
    g1[2] = (int)(((cols >> 16) & 0xFFFFu) | ((rows & 0xFFFFu) << 16));
    g1[3] = (int)(((rows >> 16) & 0xFFFFu) | ((cols & 0xFFFFu) << 16)); // tile_dim0
    g1[4] = (int)(rows & 0xFFFFu);              // tile_dim1 (tile_dim2 = 0)
    g1[5] = (int)ld;                            // tensor_dim0_stride[31:0]
    g1[6] = 0;
    g1[7] = 0;
    const i32x4 z4 = {0, 0, 0, 0};
    const i32x8 z8 = {0, 0, 0, 0, 0, 0, 0, 0};
    __builtin_amdgcn_tensor_load_to_lds(g0, g1, z4, z4, z8, 0);
}
#endif

// Stage one 32x64 bf16 tile (4KB) into LDS.
__device__ __forceinline__ void stage_tile_32x64(bf16_t* lds_dst,
                                                 const bf16_t* gsrc, int ld) {
#if USE_TDM
    if (threadIdx.x < 32) {  // wave 0, wave-uniform branch (EXEC all ones)
        tdm_load_2d_bf16((unsigned int)(uintptr_t)lds_dst, gsrc, 32u, 64u,
                         (unsigned int)ld);
        __builtin_amdgcn_s_wait_tensorcnt(0);
    }
#else
    const int t = threadIdx.x;          // 256 threads x 16B = 4KB
    const int r = t >> 3, c = (t & 7) * 8;
    *(bf16x8*)(lds_dst + r * 64 + c) = *(const bf16x8*)(gsrc + (size_t)r * ld + c);
#endif
}

// ---------------------------------------------------------------------------
// 1) RMSNorm -> bf16 activations
// ---------------------------------------------------------------------------
__global__ void __launch_bounds__(256)
rmsnorm_kernel(const float* __restrict__ seq, const float* __restrict__ w,
               bf16_t* __restrict__ xb) {
    const int row = blockIdx.x;
    const int tid = threadIdx.x;
    const float4 x = ((const float4*)(seq + (size_t)row * MODEL))[tid];
    float ss = x.x * x.x + x.y * x.y + x.z * x.z + x.w * x.w;
    __shared__ float red[256];
    red[tid] = ss;
    __syncthreads();
    for (int s = 128; s > 0; s >>= 1) {
        if (tid < s) red[tid] += red[tid + s];
        __syncthreads();
    }
    const float r = rsqrtf(red[0] * (1.0f / MODEL) + 1.1920928955078125e-07f);
    const float4 ww = ((const float4*)w)[tid];
    bf16_t* dst = xb + (size_t)row * MODEL + tid * 4;
    dst[0] = (bf16_t)(x.x * r * ww.x);
    dst[1] = (bf16_t)(x.y * r * ww.y);
    dst[2] = (bf16_t)(x.z * r * ww.z);
    dst[3] = (bf16_t)(x.w * r * ww.w);
}

// ---------------------------------------------------------------------------
// 2) f32 -> bf16 weight conversion
// ---------------------------------------------------------------------------
__global__ void cvt_bf16_kernel(const float* __restrict__ in,
                                bf16_t* __restrict__ out, int n) {
    for (int i = blockIdx.x * blockDim.x + threadIdx.x; i < n;
         i += gridDim.x * blockDim.x)
        out[i] = (bf16_t)in[i];
}

// ---------------------------------------------------------------------------
// 3) Gate GEMM: G = silu(X @ w1) * (X @ w3), bf16 out.
//    Block tile 128(M) x 64(N); 8 waves, wave w owns rows [w*16, w*16+16).
//    w1/w3 32x64 k-slabs staged in LDS via TDM, shared by all 8 waves.
//    Per k-step: 8KB staged -> 64 WMMAs (~128 flop per staged byte).
// ---------------------------------------------------------------------------
__global__ void __launch_bounds__(256)
ffn_gate_kernel(const bf16_t* __restrict__ xb, const bf16_t* __restrict__ w1,
                const bf16_t* __restrict__ w3, bf16_t* __restrict__ g) {
    __shared__ __align__(16) bf16_t sB1[32 * 64];
    __shared__ __align__(16) bf16_t sB3[32 * 64];
    const int lane = threadIdx.x & 31;
    const int wave = threadIdx.x >> 5;
    const int row0 = blockIdx.y * 128 + wave * 16;
    const int col0 = blockIdx.x * 64;

    CFrag acc1[4], acc3[4];
#pragma unroll
    for (int nt = 0; nt < 4; ++nt) {
        acc1[nt].v = (v8f)0.0f;
        acc3[nt].v = (v8f)0.0f;
    }

    for (int k0 = 0; k0 < MODEL; k0 += 32) {
        __syncthreads();  // previous slab fully consumed
        stage_tile_32x64(sB1, w1 + (size_t)k0 * HID + col0, HID);
        stage_tile_32x64(sB3, w3 + (size_t)k0 * HID + col0, HID);
        __syncthreads();  // slab visible to all waves

        const v16bf a =
            load_a_frag(xb + (size_t)row0 * MODEL + k0, MODEL, lane);
#pragma unroll
        for (int nt = 0; nt < 4; ++nt) {
            const v16bf b1 = load_b_frag(sB1 + nt * 16, 64, lane);
            const v16bf b3 = load_b_frag(sB3 + nt * 16, 64, lane);
            acc1[nt].v = __builtin_amdgcn_wmma_f32_16x16x32_bf16(
                false, a, false, b1, (short)0, acc1[nt].v, false, false);
            acc3[nt].v = __builtin_amdgcn_wmma_f32_16x16x32_bf16(
                false, a, false, b3, (short)0, acc3[nt].v, false, false);
        }
    }

    const int m = lane & 15, half = lane >> 4;
#pragma unroll
    for (int nt = 0; nt < 4; ++nt) {
#pragma unroll
        for (int i = 0; i < 8; ++i) {
            const float a1 = acc1[nt].f[i];
            const float a3 = acc3[nt].f[i];
            const float val = (a1 / (1.0f + __expf(-a1))) * a3;  // silu*gate
            g[(size_t)(row0 + half * 8 + i) * HID + col0 + nt * 16 + m] =
                (bf16_t)val;
        }
    }
}

// ---------------------------------------------------------------------------
// 4) Down GEMM + residual: Hb = G @ w2 + seq (bf16 out).
//    Block tile 128(M) x 64(N), w2 slab staged in LDS via TDM.
// ---------------------------------------------------------------------------
__global__ void __launch_bounds__(256)
ffn_down_kernel(const bf16_t* __restrict__ g, const bf16_t* __restrict__ w2,
                const float* __restrict__ seq, bf16_t* __restrict__ hb) {
    __shared__ __align__(16) bf16_t sB[32 * 64];
    const int lane = threadIdx.x & 31;
    const int wave = threadIdx.x >> 5;
    const int row0 = blockIdx.y * 128 + wave * 16;
    const int col0 = blockIdx.x * 64;

    CFrag acc[4];
#pragma unroll
    for (int nt = 0; nt < 4; ++nt) acc[nt].v = (v8f)0.0f;

    for (int k0 = 0; k0 < HID; k0 += 32) {
        __syncthreads();
        stage_tile_32x64(sB, w2 + (size_t)k0 * MODEL + col0, MODEL);
        __syncthreads();

        const v16bf a = load_a_frag(g + (size_t)row0 * HID + k0, HID, lane);
#pragma unroll
        for (int nt = 0; nt < 4; ++nt) {
            const v16bf b = load_b_frag(sB + nt * 16, 64, lane);
            acc[nt].v = __builtin_amdgcn_wmma_f32_16x16x32_bf16(
                false, a, false, b, (short)0, acc[nt].v, false, false);
        }
    }

    const int m = lane & 15, half = lane >> 4;
#pragma unroll
    for (int nt = 0; nt < 4; ++nt) {
#pragma unroll
        for (int i = 0; i < 8; ++i) {
            const size_t idx =
                (size_t)(row0 + half * 8 + i) * MODEL + col0 + nt * 16 + m;
            hb[idx] = (bf16_t)(acc[nt].f[i] + seq[idx]);
        }
    }
}

// ---------------------------------------------------------------------------
// 5) KV projection: kv[b,n,t,e] = sum_d Hb[b*T+t, n*D+d] * w_kv[n,d,e]
//    Per (b,n): 512x128 @ 128x256 (K=128: direct loads, weights L2-resident).
// ---------------------------------------------------------------------------
__global__ void __launch_bounds__(256)
kv_gemm_kernel(const bf16_t* __restrict__ hb, const bf16_t* __restrict__ wkv,
               float* __restrict__ kv) {
    const int lane  = threadIdx.x & 31;
    const int gw    = blockIdx.x * 8 + (threadIdx.x >> 5);
    const int ntile = gw & 15;          // (2D)/16
    const int mtile = (gw >> 4) & 31;   // T/16
    const int head  = (gw >> 9) & 7;
    const int b     = gw >> 12;
    const int row0  = mtile * 16;
    const int col0  = ntile * 16;

    CFrag acc; acc.v = (v8f)0.0f;
    const bf16_t* abase = hb + ((size_t)b * TT + row0) * MODEL + head * DD;
    const bf16_t* bbase = wkv + (size_t)head * DD * (2 * DD) + col0;
    for (int k0 = 0; k0 < DD; k0 += 32) {
        const v16bf a = load_a_frag(abase + k0, MODEL, lane);
        const v16bf w = load_b_frag(bbase + (size_t)k0 * (2 * DD), 2 * DD, lane);
        acc.v = __builtin_amdgcn_wmma_f32_16x16x32_bf16(
            false, a, false, w, (short)0, acc.v, false, false);
    }
    const int m = lane & 15, half = lane >> 4;
#pragma unroll
    for (int i = 0; i < 8; ++i) {
        kv[(((size_t)(b * NH + head) * TT) + row0 + half * 8 + i) * (2 * DD) +
           col0 + m] = acc.f[i];
    }
}

// ---------------------------------------------------------------------------
// 6) Single-query attention per (b,n): softmax(q.K^T/sqrt(D)) V + q
// ---------------------------------------------------------------------------
__global__ void __launch_bounds__(256)
attn_kernel(const float* __restrict__ q, const float* __restrict__ kv,
            const unsigned char* __restrict__ mask, float* __restrict__ out) {
    const int bn  = blockIdx.x;   // b*NH + n
    const int b   = bn / NH;
    const int tid = threadIdx.x;
    __shared__ float sq[DD];
    __shared__ float sc[TT];
    __shared__ float red[256];

    if (tid < DD) sq[tid] = q[(size_t)bn * DD + tid];
    __syncthreads();

    const float scale = 0.088388347648318447f;  // D^-0.5, D=128
    for (int t = tid; t < TT; t += 256) {
        const float* kp = kv + ((size_t)bn * TT + t) * (2 * DD);
        float s = 0.0f;
#pragma unroll 8
        for (int d = 0; d < DD; ++d) s += sq[d] * kp[d];
        s *= scale;
        if (!mask[(size_t)b * TT + t]) s = -3.402823466e+38f;
        sc[t] = s;
    }
    __syncthreads();

    float mloc = -3.402823466e+38f;
    for (int t = tid; t < TT; t += 256) mloc = fmaxf(mloc, sc[t]);
    red[tid] = mloc;
    __syncthreads();
    for (int s = 128; s > 0; s >>= 1) {
        if (tid < s) red[tid] = fmaxf(red[tid], red[tid + s]);
        __syncthreads();
    }
    const float mx = red[0];
    __syncthreads();

    float sum = 0.0f;
    for (int t = tid; t < TT; t += 256) {
        const float e = __expf(sc[t] - mx);
        sc[t] = e;
        sum += e;
    }
    __syncthreads();
    red[tid] = sum;
    __syncthreads();
    for (int s = 128; s > 0; s >>= 1) {
        if (tid < s) red[tid] += red[tid + s];
        __syncthreads();
    }
    const float inv = 1.0f / red[0];

    if (tid < DD) {
        float acc = 0.0f;
        for (int t = 0; t < TT; ++t)
            acc += sc[t] * kv[((size_t)bn * TT + t) * (2 * DD) + DD + tid];
        out[(size_t)bn * DD + tid] = acc * inv + sq[tid];
    }
}

// ---------------------------------------------------------------------------
// Launch
// ---------------------------------------------------------------------------
extern "C" void kernel_launch(void* const* d_in, const int* in_sizes, int n_in,
                              void* d_out, int out_size, void* d_ws,
                              size_t ws_size, hipStream_t stream) {
    const float* q      = (const float*)d_in[0];
    const float* seq    = (const float*)d_in[1];
    const unsigned char* seq_mask = (const unsigned char*)d_in[2];
    const float* rms_w  = (const float*)d_in[3];
    const float* w1     = (const float*)d_in[4];
    const float* w3     = (const float*)d_in[5];
    const float* w2     = (const float*)d_in[6];
    const float* w_kv   = (const float*)d_in[7];
    float* out = (float*)d_out;

    char* ws = (char*)d_ws;
    size_t off = 0;
    auto carve = [&](size_t bytes) -> char* {
        char* p = ws + off;
        off += (bytes + 255) & ~(size_t)255;
        return p;
    };
    bf16_t* Xb   = (bf16_t*)carve((size_t)ROWS * MODEL * sizeof(bf16_t));
    bf16_t* W1b  = (bf16_t*)carve((size_t)MODEL * HID * sizeof(bf16_t));
    bf16_t* W3b  = (bf16_t*)carve((size_t)MODEL * HID * sizeof(bf16_t));
    bf16_t* W2b  = (bf16_t*)carve((size_t)HID * MODEL * sizeof(bf16_t));
    bf16_t* WKVb = (bf16_t*)carve((size_t)NH * DD * 2 * DD * sizeof(bf16_t));
    bf16_t* G    = (bf16_t*)carve((size_t)ROWS * HID * sizeof(bf16_t));
    bf16_t* Hb   = (bf16_t*)carve((size_t)ROWS * MODEL * sizeof(bf16_t));
    float*  KV   = (float*)carve((size_t)BB * NH * TT * 2 * DD * sizeof(float));
    (void)ws_size; (void)in_sizes; (void)n_in; (void)out_size;

    cvt_bf16_kernel<<<2048, 256, 0, stream>>>(w1, W1b, MODEL * HID);
    cvt_bf16_kernel<<<2048, 256, 0, stream>>>(w3, W3b, MODEL * HID);
    cvt_bf16_kernel<<<2048, 256, 0, stream>>>(w2, W2b, HID * MODEL);
    cvt_bf16_kernel<<<256, 256, 0, stream>>>(w_kv, WKVb, NH * DD * 2 * DD);

    rmsnorm_kernel<<<ROWS, 256, 0, stream>>>(seq, rms_w, Xb);

    ffn_gate_kernel<<<dim3(HID / 64, ROWS / 128), 256, 0, stream>>>(Xb, W1b,
                                                                    W3b, G);
    ffn_down_kernel<<<dim3(MODEL / 64, ROWS / 128), 256, 0, stream>>>(G, W2b,
                                                                      seq, Hb);
    kv_gemm_kernel<<<(BB * NH * 32 * 16) / 8, 256, 0, stream>>>(Hb, WKVb, KV);
    attn_kernel<<<BB * NH, 256, 0, stream>>>(q, KV, seq_mask, out);
}
